// PrunedQwen3MoeSparseMoeBlock_45268955300373
// MI455X (gfx1250) — compile-verified, matrix operand order checked by reading, not run on hardware
//
#include <hip/hip_runtime.h>

// ---------------------------------------------------------------------------
// Pruned Qwen3 MoE block for MI455X (gfx1250, wave32, WMMA).
//   - bf16 WMMA 16x16x32 (fp32 accum) for both expert GEMMs (compute-bound:
//     ~620 GFLOP vs 27us HBM floor at 23.3 TB/s).
//   - If d_ws is big enough (~311MB), weights+x are pre-converted to bf16 once
//     (HBM-BW pass) so the GEMM inner loop has ZERO fp32->bf16 VALU, and x
//     tiles are staged into LDS with global_load_async_to_lds_b128 (ASYNCcnt).
//     Otherwise fall back to in-flight conversion.
//   - MTILE=64 tokens/block: each B tile feeds 4 M-tile WMMAs.
//   - Expert = slow grid dim so each expert's weights stay L2-resident
//     (192MB L2) across all of its token tiles.
//   - SiLU uses v_rcp_f32 (result is rounded to bf16 immediately, so the
//     IEEE-division expansion is wasted VALU).
// ---------------------------------------------------------------------------

#define T_TOKENS 2048
#define HID      2048
#define EORIG    128
#define EKEPT    32
#define INTER    768
#define TOPK     8

#define MTILE    64            // tokens per block
#define KS       256           // K superchunk staged in LDS
#define XS_LD    (KS + 16)     // pad: stride % 64 dwords != 0 -> fewer bank conflicts
#define ACT_LD   (INTER + 16)

typedef __bf16 bf16_t;
typedef __attribute__((ext_vector_type(4)))  float  v4f;
typedef __attribute__((ext_vector_type(8)))  float  v8f;
typedef __attribute__((ext_vector_type(8)))  bf16_t v8bf;
typedef __attribute__((ext_vector_type(16))) bf16_t v16bf;

static __device__ __forceinline__ v8f wmma_bf16(v16bf a, v16bf b, v8f c) {
    // D = A(16x32 bf16) x B(32x16 bf16) + C(16x16 f32)
    return __builtin_amdgcn_wmma_f32_16x16x32_bf16(
        false, a, false, b, (short)0, c, false, false);
}

// Load one 16x32 bf16 operand tile from LDS in WMMA register layout.
// lane<16: row=lr, K = kc+{0..7} and kc+16+{0..7}; lane>=16: K offset +8.
static __device__ __forceinline__ v16bf lds_load_tile(const bf16_t* s, int ld,
                                                      int rowBase, int kc,
                                                      int lr, int kh) {
    const bf16_t* p = s + (rowBase + lr) * ld + kc + kh;
    v8bf lo = *(const v8bf*)(p);
    v8bf hi = *(const v8bf*)(p + 16);
    return __builtin_shufflevector(lo, hi, 0, 1, 2, 3, 4, 5, 6, 7,
                                   8, 9, 10, 11, 12, 13, 14, 15);
}

// fp32 weight row -> bf16 WMMA operand (fallback path).
static __device__ __forceinline__ v16bf load_cvt_bf16(const float* __restrict__ p) {
    v4f f0 = *(const v4f*)(p);
    v4f f1 = *(const v4f*)(p + 4);
    v4f f2 = *(const v4f*)(p + 16);
    v4f f3 = *(const v4f*)(p + 20);
    __builtin_prefetch(p + 32, 0, 1);
    v16bf b;
#pragma unroll
    for (int i = 0; i < 4; ++i) {
        b[i]      = (bf16_t)f0[i];
        b[4 + i]  = (bf16_t)f1[i];
        b[8 + i]  = (bf16_t)f2[i];
        b[12 + i] = (bf16_t)f3[i];
    }
    return b;
}

// bf16 weight row -> WMMA operand, no conversion (pre-converted path).
static __device__ __forceinline__ v16bf load_bf16_direct(const bf16_t* __restrict__ p) {
    v8bf lo = *(const v8bf*)(p);
    v8bf hi = *(const v8bf*)(p + 16);
    __builtin_prefetch(p + 32, 0, 1);
    return __builtin_shufflevector(lo, hi, 0, 1, 2, 3, 4, 5, 6, 7,
                                   8, 9, 10, 11, 12, 13, 14, 15);
}

static __device__ __forceinline__ void atomic_add_f32(float* p, float v) {
    __hip_atomic_fetch_add(p, v, __ATOMIC_RELAXED, __HIP_MEMORY_SCOPE_AGENT);
}

// Fast SiLU: x * sigmoid(x) with v_exp_f32 + v_rcp_f32 (result feeds a bf16
// round, so a correctly-rounded divide is pointless VALU).
static __device__ __forceinline__ float silu_fast(float g) {
    return g * __builtin_amdgcn_rcpf(1.0f + __expf(-g));
}

// 16B/lane async copy global->LDS (gfx1250 GLOBAL_LOAD_ASYNC_TO_LDS_B128,
// GV mode: per-lane LDS-offset VGPR + 64-bit address VGPR pair, ASYNCcnt).
static __device__ __forceinline__ void async_copy_b128(const bf16_t* gptr,
                                                       bf16_t* lptr) {
    // Low 32 bits of a generic shared pointer are the LDS byte offset.
    unsigned lds_off = (unsigned)(unsigned long long)(uintptr_t)lptr;
    unsigned long long ga = (unsigned long long)(uintptr_t)gptr;
    asm volatile("global_load_async_to_lds_b128 %0, %1, off"
                 :: "v"(lds_off), "v"(ga) : "memory");
}
static __device__ __forceinline__ void async_wait0() {
    asm volatile("s_wait_asynccnt 0x0" ::: "memory");
}

// ---------------------------------------------------------------------------
// fp32 -> bf16 bulk convert (one HBM-bandwidth pass; n multiple of 8).
// ---------------------------------------------------------------------------
__global__ __launch_bounds__(256) void cvt_bf16_kernel(
    const float* __restrict__ src, bf16_t* __restrict__ dst, long long n)
{
    long long i = ((long long)blockIdx.x * blockDim.x + threadIdx.x) * 8;
    if (i + 8 <= n) {
        v4f a = *(const v4f*)(src + i);
        v4f b = *(const v4f*)(src + i + 4);
        v8bf o;
#pragma unroll
        for (int j = 0; j < 4; ++j) { o[j] = (bf16_t)a[j]; o[4 + j] = (bf16_t)b[j]; }
        *(v8bf*)(dst + i) = o;
    }
}

// ---------------------------------------------------------------------------
// Router: logits -> mask pruned -> top-8 -> softmax over the 8 (the full
// softmax Z cancels in top_val/sum(top_val)) -> dense combine [T, EKEPT].
// ---------------------------------------------------------------------------
__global__ __launch_bounds__(128) void router_kernel(
    const float* __restrict__ x,       // [T, H]
    const float* __restrict__ gw,      // [EORIG, H]
    const int*   __restrict__ o2n,     // [EORIG]
    float*       __restrict__ combine) // [T, EKEPT]
{
    __shared__ float xrow[HID];
    __shared__ float logits[EORIG];
    const int t   = blockIdx.x;
    const int tid = threadIdx.x;

    for (int k = tid * 4; k < HID; k += 128 * 4)
        *(v4f*)(xrow + k) = *(const v4f*)(x + (size_t)t * HID + k);
    __syncthreads();

    float acc = 0.f;
    const float* wr = gw + (size_t)tid * HID;
    for (int k = 0; k < HID; k += 4) {
        v4f wv = *(const v4f*)(wr + k);
        acc += xrow[k] * wv[0] + xrow[k + 1] * wv[1]
             + xrow[k + 2] * wv[2] + xrow[k + 3] * wv[3];
    }
    logits[tid] = (o2n[tid] < 0) ? -__builtin_inff() : acc;
    if (tid < EKEPT) combine[(size_t)t * EKEPT + tid] = 0.0f;
    __syncthreads();

    if (tid == 0) {
        float vals[TOPK]; int ids[TOPK];
#pragma unroll
        for (int s = 0; s < TOPK; ++s) {
            float m = -__builtin_inff(); int mi = 0;
            for (int e = 0; e < EORIG; ++e) {
                float v = logits[e];
                if (v > m) { m = v; mi = e; }
            }
            vals[s] = m; ids[s] = mi;
            logits[mi] = -__builtin_inff();
        }
        const float mx = vals[0];
        float w8[TOPK], sum = 0.f;
#pragma unroll
        for (int s = 0; s < TOPK; ++s) { w8[s] = __expf(vals[s] - mx); sum += w8[s]; }
        const float inv = __builtin_amdgcn_rcpf(sum);
#pragma unroll
        for (int s = 0; s < TOPK; ++s)
            combine[(size_t)t * EKEPT + o2n[ids[s]]] = w8[s] * inv;
    }
}

// ---------------------------------------------------------------------------
// Expert kernel: block = (expert e, 64-token tile). 8 waves.
// Phase 1: H1 = X @ gate_up[e]^T, fused SiLU(g)*u -> bf16 act tile in LDS.
// Phase 2: out += cw * (act @ down[e]^T), atomically into d_out.
// PRE=true: bf16 weights/x from workspace, async-LDS x staging, no cvt.
// ---------------------------------------------------------------------------
template <bool PRE>
__global__ __launch_bounds__(256) void moe_expert_kernel(
    const float*  __restrict__ xf,      // [T, H] fp32
    const bf16_t* __restrict__ xb,      // [T, H] bf16 (PRE)
    const float*  __restrict__ gupf,    // [E, 2I, H] fp32
    const bf16_t* __restrict__ gupb,    // (PRE)
    const float*  __restrict__ downf,   // [E, H, I] fp32
    const bf16_t* __restrict__ downb,   // (PRE)
    const float*  __restrict__ combine, // [T, E]
    float*        __restrict__ out)     // [T, H]
{
    __shared__ bf16_t Xs[MTILE * XS_LD];     // 34 KB  x superchunk (bf16)
    __shared__ bf16_t Acts[MTILE * ACT_LD];  // 98 KB  silu(g)*u (bf16)
    __shared__ float  Cw[MTILE];

    const int e    = blockIdx.y;
    const int t0   = blockIdx.x * MTILE;
    const int tid  = threadIdx.x;
    const int lane = tid & 31;
    const int w    = tid >> 5;       // wave 0..7
    const int lr   = lane & 15;
    const int lh   = lane >> 4;
    const int kh   = lh * 8;

    if (tid < MTILE) Cw[tid] = combine[(size_t)(t0 + tid) * EKEPT + e];

    const float*  gupf_e  = gupf  + (size_t)e * (2 * INTER) * HID;
    const bf16_t* gupb_e  = gupb  + (size_t)e * (2 * INTER) * HID;
    const float*  downf_e = downf + (size_t)e * HID * INTER;
    const bf16_t* downb_e = downb + (size_t)e * HID * INTER;

    // ---------------- Phase 1: gate_up GEMM + SiLU fuse -------------------
    // 48 column-blocks of 16 over I=768; wave w owns cb in [w*6, w*6+6).
    for (int cbi = 0; cbi < 6; ++cbi) {
        const int cb = w * 6 + cbi;
        v8f accg[4] = {{0}};
        v8f accu[4] = {{0}};

        for (int ks = 0; ks < HID / KS; ++ks) {
            __syncthreads();
            if constexpr (PRE) {
                // async stage: 64 rows x 256 bf16 = 2048 x 16B chunks,
                // 8 chunks per thread, ASYNCcnt-fenced.
                const int r  = tid >> 2;
                const int c0 = (tid & 3) * 64;
                const bf16_t* gx = xb + (size_t)(t0 + r) * HID + ks * KS + c0;
                bf16_t* dst = Xs + r * XS_LD + c0;
#pragma unroll
                for (int j = 0; j < 8; ++j)
                    async_copy_b128(gx + j * 8, dst + j * 8);
                async_wait0();
            } else {
                // load fp32, convert, ds-store bf16.
                const int r  = tid >> 2;
                const int c0 = (tid & 3) * 64;
                const float* gx = xf + (size_t)(t0 + r) * HID + ks * KS + c0;
                bf16_t* dst = Xs + r * XS_LD + c0;
#pragma unroll
                for (int j = 0; j < 8; ++j) {
                    v4f f0 = *(const v4f*)(gx + j * 8);
                    v4f f1 = *(const v4f*)(gx + j * 8 + 4);
                    v8bf o;
#pragma unroll
                    for (int i = 0; i < 4; ++i) {
                        o[i] = (bf16_t)f0[i]; o[4 + i] = (bf16_t)f1[i];
                    }
                    *(v8bf*)(dst + j * 8) = o;
                }
            }
            __syncthreads();

            for (int kc = 0; kc < KS; kc += 32) {
                v16bf a[4];
#pragma unroll
                for (int m = 0; m < 4; ++m)
                    a[m] = lds_load_tile(Xs, XS_LD, 16 * m, kc, lr, kh);
                const int kg   = ks * KS + kc + kh;
                const int gRow = cb * 16 + lr;
                v16bf bg, bu;
                if constexpr (PRE) {
                    bg = load_bf16_direct(gupb_e + (size_t)gRow * HID + kg);
                    bu = load_bf16_direct(gupb_e + (size_t)(INTER + gRow) * HID + kg);
                } else {
                    bg = load_cvt_bf16(gupf_e + (size_t)gRow * HID + kg);
                    bu = load_cvt_bf16(gupf_e + (size_t)(INTER + gRow) * HID + kg);
                }
#pragma unroll
                for (int m = 0; m < 4; ++m) {
                    accg[m] = wmma_bf16(a[m], bg, accg[m]);
                    accu[m] = wmma_bf16(a[m], bu, accu[m]);
                }
            }
        }

        // SiLU(gate) * up -> Acts (bf16). C/D: row = 16*m + 8*lh + j, col = lr.
        const int col = cb * 16 + lr;
#pragma unroll
        for (int m = 0; m < 4; ++m) {
#pragma unroll
            for (int j = 0; j < 8; ++j) {
                const float g = accg[m][j];
                const float u = accu[m][j];
                const int row = 16 * m + 8 * lh + j;
                Acts[row * ACT_LD + col] = (bf16_t)(silu_fast(g) * u);
            }
        }
    }
    __syncthreads();

    // ---------------- Phase 2: down GEMM + weighted accumulate -------------
    for (int nb = w; nb < HID / 16; nb += 8) {
        v8f acc[4] = {{0}};
        const int nRow = nb * 16 + lr;
        for (int kc = 0; kc < INTER; kc += 32) {
            v16bf a[4];
#pragma unroll
            for (int m = 0; m < 4; ++m)
                a[m] = lds_load_tile(Acts, ACT_LD, 16 * m, kc, lr, kh);
            v16bf bd;
            if constexpr (PRE)
                bd = load_bf16_direct(downb_e + (size_t)nRow * INTER + kc + kh);
            else
                bd = load_cvt_bf16(downf_e + (size_t)nRow * INTER + kc + kh);
#pragma unroll
            for (int m = 0; m < 4; ++m)
                acc[m] = wmma_bf16(a[m], bd, acc[m]);
        }
        const int col = nb * 16 + lr;
#pragma unroll
        for (int m = 0; m < 4; ++m) {
#pragma unroll
            for (int j = 0; j < 8; ++j) {
                const int row = 16 * m + 8 * lh + j;
                atomic_add_f32(out + (size_t)(t0 + row) * HID + col,
                               acc[m][j] * Cw[row]);
            }
        }
    }
}

__global__ void zero_kernel(float* __restrict__ p, int n) {
    int i = blockIdx.x * blockDim.x + threadIdx.x;
    if (i < n) p[i] = 0.0f;
}

extern "C" void kernel_launch(void* const* d_in, const int* in_sizes, int n_in,
                              void* d_out, int out_size, void* d_ws, size_t ws_size,
                              hipStream_t stream) {
    const float* x    = (const float*)d_in[0];   // hidden_states [1,2048,2048]
    const float* gw   = (const float*)d_in[1];   // gate_weight   [128,2048]
    const float* gup  = (const float*)d_in[2];   // gate_up_proj  [32,1536,2048]
    const float* down = (const float*)d_in[3];   // down_proj     [32,2048,768]
    const int*   o2n  = (const int*)d_in[4];     // old_to_new    [128]
    float* out = (float*)d_out;                  // [2048, 2048] fp32

    (void)in_sizes; (void)n_in; (void)out_size;

    // Workspace layout: [combine f32 | x_bf16 | gup_bf16 | down_bf16]
    const long long nX    = (long long)T_TOKENS * HID;
    const long long nGup  = (long long)EKEPT * 2 * INTER * HID;
    const long long nDown = (long long)EKEPT * HID * INTER;
    const size_t combine_bytes = (size_t)T_TOKENS * EKEPT * sizeof(float);
    const size_t xb_off   = combine_bytes;                  // 256 KB
    const size_t gup_off  = xb_off + (size_t)nX * 2;        // +8 MB
    const size_t down_off = gup_off + (size_t)nGup * 2;     // +192 MB
    const size_t need     = down_off + (size_t)nDown * 2;   // +96 MB ~ 311 MB

    float*  combine = (float*)d_ws;
    bf16_t* xb      = (bf16_t*)((char*)d_ws + xb_off);
    bf16_t* gupb    = (bf16_t*)((char*)d_ws + gup_off);
    bf16_t* downb   = (bf16_t*)((char*)d_ws + down_off);

    const int n = T_TOKENS * HID;
    zero_kernel<<<(n + 1023) / 1024, 1024, 0, stream>>>(out, n);
    router_kernel<<<T_TOKENS, 128, 0, stream>>>(x, gw, o2n, combine);

    dim3 grid(T_TOKENS / MTILE, EKEPT);   // expert = slow dim -> L2 residency
    if (ws_size >= need) {
        // One HBM-BW pass to bf16: removes all fp32->bf16 VALU from the GEMM
        // inner loops and halves weight bytes.
        cvt_bf16_kernel<<<(int)((nX    / 8 + 255) / 256), 256, 0, stream>>>(x,    xb,    nX);
        cvt_bf16_kernel<<<(int)((nGup  / 8 + 255) / 256), 256, 0, stream>>>(gup,  gupb,  nGup);
        cvt_bf16_kernel<<<(int)((nDown / 8 + 255) / 256), 256, 0, stream>>>(down, downb, nDown);
        moe_expert_kernel<true><<<grid, 256, 0, stream>>>(
            x, xb, gup, gupb, down, downb, combine, out);
    } else {
        moe_expert_kernel<false><<<grid, 256, 0, stream>>>(
            x, xb, gup, gupb, down, downb, combine, out);
    }
}
